// SpatialAttention_21208548508376
// MI455X (gfx1250) — matrix operand order repaired and checked by reading, not compiled, not run on hardware
//
#include <hip/hip_runtime.h>
#include <hip/hip_bf16.h>
#include <math.h>

typedef __attribute__((ext_vector_type(16))) __bf16 v16bf;
typedef __attribute__((ext_vector_type(8)))  __bf16 v8bf;
typedef __attribute__((ext_vector_type(4)))  __bf16 v4bf;
typedef __attribute__((ext_vector_type(8)))  float  v8f;
typedef __attribute__((ext_vector_type(4)))  float  v4f;

#define CAT16(a, b) __builtin_shufflevector(a, b, 0,1,2,3,4,5,6,7,8,9,10,11,12,13,14,15)

#if __has_builtin(__builtin_amdgcn_tanhf)
#define TANHF(x) __builtin_amdgcn_tanhf(x)   // v_tanh_f32 (gfx1250 trans op)
#else
#define TANHF(x) tanhf(x)
#endif

// Sizes from the reference
#define BB 32
#define TT 128
#define NN 64
#define DF 1280
#define DQ 512
#define UU 512

// ---------------------------------------------------------------------------
// Generic tiled GEMM: C[M,N] = A[M,K] * B[K,N] (+bias[N]), fp32 in/out,
// bf16 WMMA (v_wmma_f32_16x16x32_bf16), batched via blockIdx.z.
// Block = 256 threads (8 waves). C tile 128x64; wave w owns a 32x32 subtile.
// Software-pipelined: next K-slab's global loads are issued before the WMMAs
// of the current slab, so global latency hides under matrix work.
// ---------------------------------------------------------------------------
__global__ __launch_bounds__(256)
void wmma_gemm_bf16(const float* __restrict__ A, const float* __restrict__ Bm,
                    const float* __restrict__ bias, float* __restrict__ C,
                    int K, int lda, int ldb, int ldc,
                    long sA, long sB, long sC)
{
    __shared__ __align__(16) __bf16 lA[128][40];  // [m][k], padded stride
    __shared__ __align__(16) __bf16 lB[64][40];   // transposed: [n][k]

    const int t    = threadIdx.x;
    const int w    = t >> 5;
    const int lane = t & 31;
    const int hl   = lane >> 4;     // half of wave
    const int ln   = lane & 15;
    const int tileN = blockIdx.x * 64;
    const int tileM = blockIdx.y * 128;

    A  += (long)blockIdx.z * sA;
    Bm += (long)blockIdx.z * sB;
    C  += (long)blockIdx.z * sC;

    const int rb = (w & 3) * 32;    // wave's row base in tile
    const int cb = (w >> 2) * 32;   // wave's col base in tile

    // ---- per-thread global pointers (advance per K-slab, no per-iter mul) --
    const float* aP[4];
    __bf16*      sA_[4];
    #pragma unroll
    for (int rep = 0; rep < 4; ++rep) {
        const int idx = t + rep * 256;           // A tile: 128 x 32
        aP[rep]  = A + (long)(tileM + (idx >> 3)) * lda + (idx & 7) * 4;
        sA_[rep] = &lA[idx >> 3][(idx & 7) * 4];
    }
    const float* bP[2];
    __bf16*      sB_[2];
    #pragma unroll
    for (int rep = 0; rep < 2; ++rep) {
        const int idx = t + rep * 256;           // B tile: 32 x 64, transposed
        bP[rep]  = Bm + (long)(idx >> 4) * ldb + tileN + (idx & 15) * 4;
        sB_[rep] = &lB[(idx & 15) * 4][idx >> 4];
    }

    // ---- prologue: stage slab 0 into registers ----
    v4f aR[4], bR[2];
    #pragma unroll
    for (int rep = 0; rep < 4; ++rep) { aR[rep] = *(const v4f*)aP[rep]; aP[rep] += 32; }
    #pragma unroll
    for (int rep = 0; rep < 2; ++rep) { bR[rep] = *(const v4f*)bP[rep]; bP[rep] += (long)32 * ldb; }

    v8f acc[2][2] = {};

    for (int kk = 0; kk < K; kk += 32) {
        // ---- convert staged registers -> bf16 LDS ----
        #pragma unroll
        for (int rep = 0; rep < 4; ++rep) {
            v4bf ab;
            ab.x = (__bf16)aR[rep].x; ab.y = (__bf16)aR[rep].y;
            ab.z = (__bf16)aR[rep].z; ab.w = (__bf16)aR[rep].w;
            *(v4bf*)sA_[rep] = ab;               // ds_store_b64
        }
        #pragma unroll
        for (int rep = 0; rep < 2; ++rep) {
            __bf16* p = sB_[rep];                // 4 cols -> strided b16 stores
            p[0]      = (__bf16)bR[rep].x;
            p[40]     = (__bf16)bR[rep].y;
            p[80]     = (__bf16)bR[rep].z;
            p[120]    = (__bf16)bR[rep].w;
        }
        __syncthreads();

        // ---- issue next slab's global loads; latency hides under WMMAs ----
        if (kk + 32 < K) {
            #pragma unroll
            for (int rep = 0; rep < 4; ++rep) { aR[rep] = *(const v4f*)aP[rep]; aP[rep] += 32; }
            #pragma unroll
            for (int rep = 0; rep < 2; ++rep) { bR[rep] = *(const v4f*)bP[rep]; bP[rep] += (long)32 * ldb; }
            if (kk + 64 < K) {                   // prefetch two slabs ahead
                __builtin_prefetch(aP[0], 0, 0);
                __builtin_prefetch(bP[0], 0, 0);
            }
        }

        // ---- A fragments: 16x32 bf16; lane ln = row, K split by half ----
        //      VGPR j<8 -> K = hl*8 + j ; j>=8 -> K = 16 + hl*8 + (j-8)
        v16bf af[2];
        #pragma unroll
        for (int ai = 0; ai < 2; ++ai) {
            const __bf16* pa = &lA[rb + ai * 16 + ln][0];
            af[ai] = CAT16(*(const v8bf*)(pa + hl * 8),
                           *(const v8bf*)(pa + 16 + hl * 8));
        }
        // ---- B fragments: 32x16 bf16; lane ln = col, K split by half like A ----
        v16bf bfr[2];
        #pragma unroll
        for (int bi = 0; bi < 2; ++bi) {
            const __bf16* pb = &lB[cb + bi * 16 + ln][0];
            bfr[bi] = CAT16(*(const v8bf*)(pb + hl * 8), *(const v8bf*)(pb + 16 + hl * 8));
        }
        #pragma unroll
        for (int ai = 0; ai < 2; ++ai)
            #pragma unroll
            for (int bi = 0; bi < 2; ++bi)
                acc[ai][bi] = __builtin_amdgcn_wmma_f32_16x16x32_bf16(
                    false, af[ai], false, bfr[bi], (short)0, acc[ai][bi],
                    false, false);
        __syncthreads();
    }

    // ---- epilogue: C/D layout VGPR e -> row = e + hl*8, col = ln ----
    #pragma unroll
    for (int bi = 0; bi < 2; ++bi) {
        const int gcol = tileN + cb + bi * 16 + ln;
        const float bv = bias ? bias[gcol] : 0.0f;
        #pragma unroll
        for (int ai = 0; ai < 2; ++ai) {
            #pragma unroll
            for (int e = 0; e < 8; ++e) {
                const int grow = tileM + rb + ai * 16 + hl * 8 + e;
                C[(long)grow * ldc + gcol] = acc[ai][bi][e] + bv;
            }
        }
    }
}

// ---------------------------------------------------------------------------
// Fused score + softmax per (b,t) row:
//   score[n] = V_b + sum_u V_w[u] * tanh(qp[bt,u] + fp[b,n,u])
//   alpha[bt,:] = softmax_n(score)
// One block per bt; q-row and V_w staged in LDS. Wave w computes n = w*8..+7;
// native v_tanh_f32 via __builtin_amdgcn_tanhf.
// ---------------------------------------------------------------------------
__global__ __launch_bounds__(256)
void score_softmax_kernel(const float* __restrict__ qp,
                          const float* __restrict__ fp,
                          const float* __restrict__ Vw,
                          const float* __restrict__ Vb,
                          float* __restrict__ alpha)
{
    __shared__ float sQ[UU];
    __shared__ float sV[UU];
    __shared__ float sS[NN];

    const int bt = blockIdx.x;          // 0 .. B*T-1
    const int b  = bt >> 7;             // / T
    const int t  = threadIdx.x;

    sQ[t]       = qp[(long)bt * UU + t];
    sQ[t + 256] = qp[(long)bt * UU + t + 256];
    sV[t]       = Vw[t];
    sV[t + 256] = Vw[t + 256];
    __syncthreads();

    const int w    = t >> 5;
    const int lane = t & 31;

    #pragma unroll
    for (int j = 0; j < 8; ++j) {
        const int n = w * 8 + j;
        const float* f = fp + ((long)(b * NN + n)) * UU;
        float s = 0.0f;
        #pragma unroll
        for (int i = 0; i < UU / 32; ++i) {
            const int u = i * 32 + lane;
            s = fmaf(sV[u], TANHF(sQ[u] + f[u]), s);
        }
        #pragma unroll
        for (int off = 16; off > 0; off >>= 1)
            s += __shfl_xor(s, off, 32);
        if (lane == 0)
            sS[n] = s + Vb[0];
    }
    __syncthreads();

    if (w == 0) {   // softmax over 64 values, 2 per lane
        float v0 = sS[lane], v1 = sS[lane + 32];
        float m = fmaxf(v0, v1);
        #pragma unroll
        for (int off = 16; off > 0; off >>= 1)
            m = fmaxf(m, __shfl_xor(m, off, 32));
        float e0 = __expf(v0 - m);
        float e1 = __expf(v1 - m);
        float sum = e0 + e1;
        #pragma unroll
        for (int off = 16; off > 0; off >>= 1)
            sum += __shfl_xor(sum, off, 32);
        const float inv = 1.0f / sum;
        alpha[(long)bt * NN + lane]      = e0 * inv;
        alpha[(long)bt * NN + lane + 32] = e1 * inv;
    }
}

// ---------------------------------------------------------------------------
extern "C" void kernel_launch(void* const* d_in, const int* in_sizes, int n_in,
                              void* d_out, int out_size, void* d_ws, size_t ws_size,
                              hipStream_t stream)
{
    const float* img = (const float*)d_in[0];   // [B,N,DF]
    const float* qry = (const float*)d_in[1];   // [B,T,DQ]
    const float* Wf  = (const float*)d_in[2];   // [DF,U]
    const float* bft = (const float*)d_in[3];   // [U]
    const float* Wq  = (const float*)d_in[4];   // [DQ,U]
    const float* bq  = (const float*)d_in[5];   // [U]
    const float* Vw  = (const float*)d_in[6];   // [U,1]
    const float* Vb  = (const float*)d_in[7];   // [1]

    float* out = (float*)d_out;
    float* ws  = (float*)d_ws;

    float* fp = ws;                                   // B*N*U  = 1,048,576 floats
    float* qp = fp + (long)BB * NN * UU;              // B*T*U  = 2,097,152 floats

    float* ctx   = out;                               // [B,T,DF]
    float* alpha = out + (long)BB * TT * DF;          // [B,T,N]

    // q_proj = query @ W_query + b_query : M=4096, N=512, K=512
    wmma_gemm_bf16<<<dim3(UU / 64, (BB * TT) / 128, 1), 256, 0, stream>>>(
        qry, Wq, bq, qp, DQ, DQ, UU, UU, 0, 0, 0);

    // f_proj = img_proj @ W_feat + b_feat : M=2048, N=512, K=1280
    wmma_gemm_bf16<<<dim3(UU / 64, (BB * NN) / 128, 1), 256, 0, stream>>>(
        img, Wf, bft, fp, DF, DF, UU, UU, 0, 0, 0);

    // alpha = softmax_n( V_b + tanh(qp + fp) . V_w )  -- fused
    score_softmax_kernel<<<BB * TT, 256, 0, stream>>>(qp, fp, Vw, Vb, alpha);

    // context[b] = alpha[b] @ img_proj[b] : batched M=128, N=1280, K=64
    wmma_gemm_bf16<<<dim3(DF / 64, TT / 128, BB), 256, 0, stream>>>(
        alpha, img, nullptr, ctx, NN, NN, DF, DF,
        (long)TT * NN, (long)NN * DF, (long)TT * DF);
}
